// MG_77618648973417
// MI455X (gfx1250) — matrix operand order repaired and checked by reading, not compiled, not run on hardware
//
#include <hip/hip_runtime.h>

typedef __attribute__((ext_vector_type(16))) _Float16 v16h;
typedef __attribute__((ext_vector_type(8)))  _Float16 v8h;
typedef __attribute__((ext_vector_type(8)))  float    v8f;

#define N_NODES 50000
#define N_EDGES 600000
#define DIM     128
#define HID     128
#define N_MASK  25000
#define EPS_BN   1e-5f
#define EPS_NORM 1e-12f

#if defined(__has_builtin)
#if __has_builtin(__builtin_amdgcn_global_load_async_to_lds_b128)
#define HAVE_ASYNC_LDS 1
#endif
#endif

// ---------------------------------------------------------------- utilities
__global__ void k_zero(float* __restrict__ p, long n) {
  long i = (long)blockIdx.x * blockDim.x + threadIdx.x;
  long stride = (long)gridDim.x * blockDim.x;
  for (; i < n; i += stride) p[i] = 0.0f;
}

// degree counts via hardware f32 atomics
__global__ void k_deg(const int* __restrict__ src, const int* __restrict__ dst,
                      float* __restrict__ cnt_out, float* __restrict__ cnt_in, int E) {
  int e = blockIdx.x * blockDim.x + threadIdx.x;
  if (e < E) {
    unsafeAtomicAdd(&cnt_out[src[e]], 1.0f);
    unsafeAtomicAdd(&cnt_in[dst[e]],  1.0f);
  }
}

__global__ void k_rsqrt_clip(float* __restrict__ d, int n) {
  int i = blockIdx.x * blockDim.x + threadIdx.x;
  if (i < n) d[i] = rsqrtf(fmaxf(d[i], 1.0f));
}

// Wt[n*DIM + k] = (f16) W[k*HID + n]   (column-major f16 weights for WMMA-B)
__global__ void k_f2h_t(const float* __restrict__ W, _Float16* __restrict__ Wt) {
  int i = blockIdx.x * blockDim.x + threadIdx.x;
  if (i < DIM * HID) {
    int n = i >> 7, k = i & 127;
    Wt[i] = (_Float16)W[k * HID + n];
  }
}

// Xh[i] = feat[i] * (enc ? rsqrt(deg_out[row]) : 1), converted to f16
__global__ void k_prep_xh(const float* __restrict__ feat, const float* __restrict__ rs,
                          _Float16* __restrict__ Xh, int scale_flag) {
  long i = (long)blockIdx.x * blockDim.x + threadIdx.x;
  if (i < (long)N_NODES * DIM) {
    float v = feat[i];
    if (scale_flag) v *= rs[i >> 7];   // row = i / 128
    Xh[i] = (_Float16)v;
  }
}

// zero masked rows of Xh (duplicates in mask are harmless)
__global__ void k_mask_rows(_Float16* __restrict__ Xh, const int* __restrict__ mask, int M) {
  int t = blockIdx.x * blockDim.x + threadIdx.x;
  if (t < M * 32) {
    int row  = mask[t >> 5];
    int lane = t & 31;                 // 32 lanes x 4 halfs = 128 halfs/row
    ((uint2*)(Xh + (size_t)row * DIM))[lane] = make_uint2(0u, 0u);
  }
}

// ---------------------------------------------------------------- WMMA GEMM
// Q[N x 128] = Xh[N x 128] @ W[128 x 128] (Wt is col-major f16)
// one block = 16 output rows, 8 waves x 16 cols; W + X tile staged in LDS
__global__ __launch_bounds__(256) void k_gemm(const _Float16* __restrict__ Xh,
                                              const _Float16* __restrict__ Wt,
                                              float* __restrict__ Q) {
  __shared__ _Float16 sWt[HID * DIM];  // 32 KB, col-major: sWt[n*DIM + k]
  __shared__ _Float16 sX[16 * DIM];    //  4 KB, row-major
  const int tid = threadIdx.x;
  const int m0  = blockIdx.x * 16;

  const uint4* gW = (const uint4*)Wt;                       // 2048 x uint4
  const uint4* gX = (const uint4*)(Xh + (size_t)m0 * DIM);  //  256 x uint4
  uint4*       lW = (uint4*)sWt;
  uint4*       lX = (uint4*)sX;

#ifdef HAVE_ASYNC_LDS
  // gfx1250 async DMA into LDS (ASYNCcnt-tracked), per-lane 16B transfers.
  // Builtin expects (v4i AS1* src, v4i AS3* dst, imm offset, imm cpol).
  typedef int v4i __attribute__((vector_size(16)));
  typedef __attribute__((address_space(1))) v4i* gv4_t;
  typedef __attribute__((address_space(3))) v4i* lv4_t;
#pragma unroll
  for (int r = 0; r < 8; ++r)
    __builtin_amdgcn_global_load_async_to_lds_b128((gv4_t)(gW + tid + 256 * r),
                                                   (lv4_t)(lW + tid + 256 * r), 0, 0);
  __builtin_amdgcn_global_load_async_to_lds_b128((gv4_t)(gX + tid),
                                                 (lv4_t)(lX + tid), 0, 0);
  asm volatile("s_wait_asynccnt 0x0" ::: "memory");
#else
#pragma unroll
  for (int r = 0; r < 8; ++r) lW[tid + 256 * r] = gW[tid + 256 * r];
  lX[tid] = gX[tid];
#endif
  __syncthreads();

  const int wave = tid >> 5;
  const int lane = tid & 31;
  const int hi   = lane >> 4;   // half-wave select
  const int lm   = lane & 15;
  const int n0   = wave * 16;

  const _Float16* ax = sX  + lm * DIM;          // A row M = lm
  const _Float16* bx = sWt + (n0 + lm) * DIM;   // B col N = n0+lm

  v8f c = {};
#pragma unroll
  for (int kb = 0; kb < DIM; kb += 32) {
    // A 16x32 f16 layout: lane(hi,lm) holds K = kb+hi*16 + [0..15]
    v8h a0 = *(const v8h*)(ax + kb + hi * 16);
    v8h a1 = *(const v8h*)(ax + kb + hi * 16 + 8);
    v16h a = __builtin_shufflevector(a0, a1, 0, 1, 2, 3, 4, 5, 6, 7,
                                     8, 9, 10, 11, 12, 13, 14, 15);
    // B 32x16 f16 layout: lane(hi,lm) holds K = kb+hi*16 + [0..15] of col n0+lm
    v16h b = *(const v16h*)(bx + kb + hi * 16);
    c = __builtin_amdgcn_wmma_f32_16x16x32_f16(false, a, false, b, (short)0, c,
                                               false, false);
  }
  // D layout: VGPR r -> M = r + hi*8, N = lm
#pragma unroll
  for (int r = 0; r < 8; ++r)
    Q[(size_t)(m0 + r + hi * 8) * HID + n0 + lm] = c[r];
}

// ------------------------------------------------------- edge scatter-add
// one wave per edge; lane grabs 4 floats of the source row
__global__ void k_scatter(const int* __restrict__ src, const int* __restrict__ dst,
                          const float* __restrict__ w, const float* __restrict__ Q,
                          float* __restrict__ P, int E) {
  int t = blockIdx.x * blockDim.x + threadIdx.x;
  int e = t >> 5;
  if (e >= E) return;
  int lane = t & 31;
  int s = src[e], d = dst[e];
  float4 v = ((const float4*)(Q + (size_t)s * HID))[lane];
  if (w) { float ww = w[e]; v.x *= ww; v.y *= ww; v.z *= ww; v.w *= ww; }
  float* pd = P + (size_t)d * HID + lane * 4;
  unsafeAtomicAdd(pd + 0, v.x);
  unsafeAtomicAdd(pd + 1, v.y);
  unsafeAtomicAdd(pd + 2, v.z);
  unsafeAtomicAdd(pd + 3, v.w);
}

// P = prelu(P * (enc ? rsqrt(deg_in) : 1) + b, a)   (in place)
__global__ void k_post(float* __restrict__ P, const float* __restrict__ rs_in,
                       const float* __restrict__ bias, const float* __restrict__ aPtr) {
  long i = (long)blockIdx.x * blockDim.x + threadIdx.x;
  if (i < (long)N_NODES * HID) {
    float a = aPtr[0];
    float v = P[i];
    if (rs_in) v *= rs_in[i >> 7];
    v += bias[i & 127];
    P[i] = (v >= 0.0f) ? v : a * v;
  }
}

// column sums / sumsq; thread owns a fixed column (stride keeps c constant)
__global__ __launch_bounds__(256) void k_bnsum(const float* __restrict__ P,
                                               float* __restrict__ sums, long n) {
  long i = (long)blockIdx.x * 256 + threadIdx.x;
  long stride = (long)gridDim.x * 256;
  float s = 0.0f, q = 0.0f;
  for (; i < n; i += stride) { float v = P[i]; s += v; q += v * v; }
  __shared__ float ls[256], lq[256];
  ls[threadIdx.x] = s; lq[threadIdx.x] = q;
  __syncthreads();
  if (threadIdx.x < 128) {
    s = ls[threadIdx.x] + ls[threadIdx.x + 128];
    q = lq[threadIdx.x] + lq[threadIdx.x + 128];
    unsafeAtomicAdd(&sums[threadIdx.x], s);
    unsafeAtomicAdd(&sums[128 + threadIdx.x], q);
  }
}

// O = prelu((P - mu) * rsqrt(var + eps) * gamma + beta, aact)
__global__ void k_bnapply(const float* __restrict__ P, const float* __restrict__ sums,
                          const float* __restrict__ gamma, const float* __restrict__ beta,
                          const float* __restrict__ aactPtr, float* __restrict__ O) {
  long i = (long)blockIdx.x * blockDim.x + threadIdx.x;
  if (i < (long)N_NODES * HID) {
    int c = i & 127;
    float mu  = sums[c] * (1.0f / N_NODES);
    float var = sums[128 + c] * (1.0f / N_NODES) - mu * mu;
    float v = (P[i] - mu) * rsqrtf(var + EPS_BN) * gamma[c] + beta[c];
    float a = aactPtr[0];
    O[i] = (v >= 0.0f) ? v : a * v;
  }
}

// sum over mask rows of (1 - cos(O1[m], O2[m])); one wave per mask entry
__global__ void k_sce(const float* __restrict__ O1, const float* __restrict__ O2,
                      const int* __restrict__ mask, int M, float* __restrict__ acc) {
  int t = blockIdx.x * blockDim.x + threadIdx.x;
  int e = t >> 5;
  if (e >= M) return;
  int lane = t & 31;
  int row = mask[e];
  float4 x = ((const float4*)(O1 + (size_t)row * HID))[lane];
  float4 y = ((const float4*)(O2 + (size_t)row * HID))[lane];
  float dot = x.x * y.x + x.y * y.y + x.z * y.z + x.w * y.w;
  float sx  = x.x * x.x + x.y * x.y + x.z * x.z + x.w * x.w;
  float sy  = y.x * y.x + y.y * y.y + y.z * y.z + y.w * y.w;
  for (int off = 16; off > 0; off >>= 1) {
    dot += __shfl_xor(dot, off, 32);
    sx  += __shfl_xor(sx,  off, 32);
    sy  += __shfl_xor(sy,  off, 32);
  }
  if (lane == 0) {
    float nx = fmaxf(sqrtf(sx), EPS_NORM);
    float ny = fmaxf(sqrtf(sy), EPS_NORM);
    unsafeAtomicAdd(acc, 1.0f - dot / (nx * ny));
  }
}

__global__ void k_final(const float* __restrict__ acc, float* __restrict__ out) {
  out[0] = 0.5f * (acc[0] + acc[1]) * (1.0f / (float)N_MASK);
}

// ---------------------------------------------------------------- host side
static void run_pass(const float* feat, const int* mask, const _Float16* Wt,
                     const float* bias, const float* aPtr, const float* gamma,
                     const float* beta, const float* aactPtr, bool enc,
                     const int* esrc, const int* edst, const float* ew,
                     const float* rs_out, const float* rs_in,
                     _Float16* Xh, float* Q, float* P, float* stats, float* O,
                     hipStream_t stream) {
  const long NH = (long)N_NODES * HID;
  k_prep_xh<<<25000, 256, 0, stream>>>(feat, rs_out, Xh, enc ? 1 : 0);
  if (mask)
    k_mask_rows<<<(N_MASK * 32 + 255) / 256, 256, 0, stream>>>(Xh, mask, N_MASK);
  k_gemm<<<N_NODES / 16, 256, 0, stream>>>(Xh, Wt, Q);
  k_zero<<<1024, 256, 0, stream>>>(P, NH);
  k_scatter<<<(N_EDGES * 32) / 256, 256, 0, stream>>>(esrc, edst, ew, Q, P, N_EDGES);
  k_post<<<25000, 256, 0, stream>>>(P, enc ? rs_in : nullptr, bias, aPtr);
  k_zero<<<1, 256, 0, stream>>>(stats, 2 * HID);
  k_bnsum<<<512, 256, 0, stream>>>(P, stats, NH);
  k_bnapply<<<25000, 256, 0, stream>>>(P, stats, gamma, beta, aactPtr, O);
}

extern "C" void kernel_launch(void* const* d_in, const int* in_sizes, int n_in,
                              void* d_out, int out_size, void* d_ws, size_t ws_size,
                              hipStream_t stream) {
  const int*   ei    = (const int*)d_in[0];
  const int*   dei   = (const int*)d_in[1];
  const float* ew    = (const float*)d_in[2];
  const float* feat  = (const float*)d_in[3];
  const int*   mask0 = (const int*)d_in[4];
  const int*   mask1 = (const int*)d_in[5];
  const float* W1    = (const float*)d_in[6];
  const float* b1    = (const float*)d_in[7];
  const float* a1    = (const float*)d_in[8];
  const float* g1    = (const float*)d_in[9];
  const float* be1   = (const float*)d_in[10];
  const float* aa1   = (const float*)d_in[11];
  const float* W2    = (const float*)d_in[12];
  const float* b2    = (const float*)d_in[13];
  const float* a2    = (const float*)d_in[14];
  const float* g2    = (const float*)d_in[15];
  const float* be2   = (const float*)d_in[16];
  const float* aa2   = (const float*)d_in[17];

  const long NH = (long)N_NODES * HID;
  float*    rs_out = (float*)d_ws;
  float*    rs_in  = rs_out + N_NODES;
  _Float16* Xh     = (_Float16*)(rs_in + N_NODES);
  _Float16* W1t    = Xh + NH;
  _Float16* W2t    = W1t + DIM * HID;
  float*    Q      = (float*)(W2t + DIM * HID);
  float*    P      = Q + NH;
  float*    O1     = P + NH;
  float*    O2     = O1 + NH;
  float*    stats  = O2 + NH;
  float*    acc    = stats + 2 * HID;

  const int* esrc = ei;
  const int* edst = ei + N_EDGES;
  const int* dsrc = dei;
  const int* ddst = dei + N_EDGES;

  // degrees for conv_both (edge_index only)
  k_zero<<<256, 256, 0, stream>>>(rs_out, 2L * N_NODES);
  k_deg<<<(N_EDGES + 255) / 256, 256, 0, stream>>>(esrc, edst, rs_out, rs_in, N_EDGES);
  k_rsqrt_clip<<<(2 * N_NODES + 255) / 256, 256, 0, stream>>>(rs_out, 2 * N_NODES);

  // weights to transposed f16 once
  k_f2h_t<<<(DIM * HID + 255) / 256, 256, 0, stream>>>(W1, W1t);
  k_f2h_t<<<(DIM * HID + 255) / 256, 256, 0, stream>>>(W2, W2t);

  k_zero<<<1, 32, 0, stream>>>(acc, 2);

  // loss1: h1 = enc(feat with mask0 zeroed), h2 = dec(feat)
  run_pass(feat, mask0, W1t, b1, a1, g1, be1, aa1, true,  esrc, edst, nullptr,
           rs_out, rs_in, Xh, Q, P, stats, O1, stream);
  run_pass(feat, nullptr, W2t, b2, a2, g2, be2, aa2, false, dsrc, ddst, ew,
           rs_out, rs_in, Xh, Q, P, stats, O2, stream);
  k_sce<<<(N_MASK * 32) / 256, 256, 0, stream>>>(O1, O2, mask0, N_MASK, acc);

  // loss2: h1b = enc(feat), h2b = dec(feat with mask1 zeroed)
  run_pass(feat, nullptr, W1t, b1, a1, g1, be1, aa1, true,  esrc, edst, nullptr,
           rs_out, rs_in, Xh, Q, P, stats, O1, stream);
  run_pass(feat, mask1, W2t, b2, a2, g2, be2, aa2, false, dsrc, ddst, ew,
           rs_out, rs_in, Xh, Q, P, stats, O2, stream);
  k_sce<<<(N_MASK * 32) / 256, 256, 0, stream>>>(O1, O2, mask1, N_MASK, acc + 1);

  k_final<<<1, 1, 0, stream>>>(acc, (float*)d_out);
}
